// CrossAttention_34909494182275
// MI455X (gfx1250) — matrix operand order repaired and checked by reading, not compiled, not run on hardware
//
#include <hip/hip_runtime.h>

// ---------------------------------------------------------------------------
// CDNA5 / gfx1250 cross-attention. wave32. WMMA bf16 (16x16x32), f32 accum.
// Pipeline: [cvt f32->bf16] -> [QKV GEMM] -> [fused attention] -> [out GEMM]
// ---------------------------------------------------------------------------

typedef __attribute__((ext_vector_type(16))) __bf16  v16bf;
typedef __attribute__((ext_vector_type(8)))  __bf16  bf16x8;
typedef __attribute__((ext_vector_type(8)))  float   v8f;

#define DEV __device__ __forceinline__

union Frag  { v16bf v; bf16x8 h[2]; __bf16 e[16]; };
union Pack8 { bf16x8 v; __bf16 e[8]; };

DEV void cvt4(const float4 f, __bf16* o) {
  o[0] = (__bf16)f.x; o[1] = (__bf16)f.y;
  o[2] = (__bf16)f.z; o[3] = (__bf16)f.w;
}

// 16-lane (half-wave) reductions via ds_swizzle (group-of-32 xor mode,
// and_mask=0x1f). xor masks 1,2,4,8 stay inside each 16-lane group.
DEV float red_max16(float v) {
  v = fmaxf(v, __int_as_float(__builtin_amdgcn_ds_swizzle(__float_as_int(v), (1 << 10) | 0x1f)));
  v = fmaxf(v, __int_as_float(__builtin_amdgcn_ds_swizzle(__float_as_int(v), (2 << 10) | 0x1f)));
  v = fmaxf(v, __int_as_float(__builtin_amdgcn_ds_swizzle(__float_as_int(v), (4 << 10) | 0x1f)));
  v = fmaxf(v, __int_as_float(__builtin_amdgcn_ds_swizzle(__float_as_int(v), (8 << 10) | 0x1f)));
  return v;
}
DEV float red_sum16(float v) {
  v += __int_as_float(__builtin_amdgcn_ds_swizzle(__float_as_int(v), (1 << 10) | 0x1f));
  v += __int_as_float(__builtin_amdgcn_ds_swizzle(__float_as_int(v), (2 << 10) | 0x1f));
  v += __int_as_float(__builtin_amdgcn_ds_swizzle(__float_as_int(v), (4 << 10) | 0x1f));
  v += __int_as_float(__builtin_amdgcn_ds_swizzle(__float_as_int(v), (8 << 10) | 0x1f));
  return v;
}

// ---------------------------------------------------------------------------
// Kernel 0: elementwise f32 -> bf16 (8 elements / thread, vectorized)
// ---------------------------------------------------------------------------
__global__ __launch_bounds__(256) void cvt_bf16_kernel(
    const float* __restrict__ in, __bf16* __restrict__ out, int n) {
  const int i = (blockIdx.x * 256 + threadIdx.x) * 8;
  if (i >= n) return;
  float4 a = *(const float4*)(in + i);
  float4 b = *(const float4*)(in + i + 4);
  Pack8 p;
  cvt4(a, p.e); cvt4(b, p.e + 4);
  *(bf16x8*)(out + i) = p.v;
}

// ---------------------------------------------------------------------------
// Kernel 1: QKV projection.  out_bf16 = A_bf16[16384x1024] @ W_bf16^T + bias
// vmode=0: flat i = m*1024+n  (== reshaped [h][b][t][d], identical flat index)
// vmode=1: V transposed [h][b][d][t]  (PV B-operand becomes contiguous)
// grid (16, 64), block 256 (8 waves). Wave: 32 rows x 64 cols, 8 WMMA/k-step.
// ---------------------------------------------------------------------------
__global__ __launch_bounds__(256) void qkv_gemm_kernel(
    const __bf16* __restrict__ A, const __bf16* __restrict__ W,
    const float* __restrict__ bias, __bf16* __restrict__ out, int vmode) {
  const int tid  = threadIdx.x;
  const int wv   = tid >> 5;
  const int lane = tid & 31;
  const int l16  = lane & 15;
  const int hi   = lane >> 4;
  const int m0   = blockIdx.y * 256 + wv * 32;
  const int n0   = blockIdx.x * 64;

  const __bf16* arow0 = A + (size_t)(m0 + l16) * 1024;
  const __bf16* arow1 = arow0 + 16 * 1024;

  v8f acc0[4] = {};
  v8f acc1[4] = {};
  for (int kt = 0; kt < 32; ++kt) {
    const int kb  = kt * 32 + hi * 16;   // A-frag chunk base
    const int kbb = kt * 32 + hi * 16;   // B-frag base (16 contiguous)
    __builtin_prefetch(arow0 + kb + 256, 0, 0);
    Frag a0, a1;
    a0.h[0] = *(const bf16x8*)(arow0 + kb);
    a0.h[1] = *(const bf16x8*)(arow0 + kb + 8);
    a1.h[0] = *(const bf16x8*)(arow1 + kb);
    a1.h[1] = *(const bf16x8*)(arow1 + kb + 8);
#pragma unroll
    for (int j = 0; j < 4; ++j) {
      v16bf bf = *(const v16bf*)(W + (size_t)(n0 + 16 * j + l16) * 1024 + kbb);
      acc0[j] = __builtin_amdgcn_wmma_f32_16x16x32_bf16(
          false, a0.v, false, bf, (short)0, acc0[j], false, false);
      acc1[j] = __builtin_amdgcn_wmma_f32_16x16x32_bf16(
          false, a1.v, false, bf, (short)0, acc1[j], false, false);
    }
  }
#pragma unroll
  for (int half = 0; half < 2; ++half) {
    const v8f* acc = half ? acc1 : acc0;
    const int mb = m0 + half * 16;
#pragma unroll
    for (int j = 0; j < 4; ++j) {
#pragma unroll
      for (int r = 0; r < 8; ++r) {
        const int mrow = mb + r + 8 * hi;           // C/D: M = r + 8*(lane>>4)
        const int ncol = n0 + 16 * j + l16;         //      N = lane&15
        const float val = acc[j][r] + bias[ncol];
        const unsigned int i = (unsigned)mrow * 1024u + (unsigned)ncol;
        unsigned int idx;
        if (vmode) {  // [h][b][t][d] -> [h][b][d][t]
          unsigned int hh = i >> 22, bp = (i >> 18) & 15u;
          unsigned int tp = (i >> 8) & 1023u, dd = i & 255u;
          idx = (((hh * 16u + bp) * 256u + dd) << 10) + tp;
        } else {
          idx = i;
        }
        out[idx] = (__bf16)val;
      }
    }
  }
}

// ---------------------------------------------------------------------------
// Kernel 2: attention for one (h, b, 16-query tile).
// S[16][1024] f32 (64KB):  S = (Q K^T)/32 + adj ; softmax writes normalized
// P into Pb[16][1024] bf16 (32KB) ; O = P V with WMMA -> temp bf16.
// grid (64, 16, 4), block 256 (8 waves). 96KB LDS (<< 320KB/WGP on CDNA5).
// ---------------------------------------------------------------------------
__global__ __launch_bounds__(256) void attn_kernel(
    const __bf16* __restrict__ Qb, const __bf16* __restrict__ Kb,
    const __bf16* __restrict__ Vtb, const float* __restrict__ adj,
    __bf16* __restrict__ tempb) {
  __shared__ float  S[16][1024];   // 64 KB
  __shared__ __bf16 Pb[16][1024];  // 32 KB

  const int h  = blockIdx.z;
  const int b  = blockIdx.y;
  const int q0 = blockIdx.x * 16;
  const int tid  = threadIdx.x;
  const int wv   = tid >> 5;
  const int lane = tid & 31;
  const int l16  = lane & 15;
  const int hi   = lane >> 4;

  const __bf16* Q = Qb  + ((size_t)((h * 16 + b) * 1024 + q0) << 8);
  const __bf16* K = Kb  + ((size_t)((h * 16 + b) * 1024) << 8);
  const __bf16* V = Vtb + ((size_t)((h * 16 + b) * 256) << 10);

  // Resident Q fragments: 8 d-steps of 32 (dh = 256).
  const __bf16* qrow = Q + (l16 << 8);
  v16bf qf[8];
#pragma unroll
  for (int ds = 0; ds < 8; ++ds) {
    const int kb = ds * 32 + hi * 16;
    Frag t;
    t.h[0] = *(const bf16x8*)(qrow + kb);
    t.h[1] = *(const bf16x8*)(qrow + kb + 8);
    qf[ds] = t.v;
  }

  const float norm = 0.03125f;  // 1/sqrt(1024)

  // ---- scores: each wave covers key-tiles kt = wv, wv+8, ... (wave-uniform)
  for (int kt = wv; kt < 64; kt += 8) {
    v8f acc = {};
    const __bf16* krow = K + ((size_t)(kt * 16 + l16) << 8) + hi * 16;
#pragma unroll
    for (int ds = 0; ds < 8; ++ds) {
      v16bf bf = *(const v16bf*)(krow + ds * 32);
      acc = __builtin_amdgcn_wmma_f32_16x16x32_bf16(
          false, qf[ds], false, bf, (short)0, acc, false, false);
    }
#pragma unroll
    for (int r = 0; r < 8; ++r) {
      const int m = r + 8 * hi;
      const int c = kt * 16 + l16;
      S[m][c] = acc[r] * norm + adj[(size_t)((b << 10) + q0 + m) * 1024 + c];
    }
  }
  __syncthreads();

  // ---- softmax: 16 threads per row; ds_swizzle reductions; write bf16 P
  {
    const int r  = tid >> 4;
    const int c0 = tid & 15;
    float mx = -3.402823466e38f;
    for (int c = c0; c < 1024; c += 16) mx = fmaxf(mx, S[r][c]);
    mx = red_max16(mx);
    float sum = 0.f;
    for (int c = c0; c < 1024; c += 16) {
      float e = __expf(S[r][c] - mx);
      S[r][c] = e;
      sum += e;
    }
    sum = red_sum16(sum);
    const float inv = 1.0f / sum;
    for (int c = c0; c < 1024; c += 16) Pb[r][c] = (__bf16)(S[r][c] * inv);
  }
  __syncthreads();

  // ---- O = P @ V : wave handles d-tiles (wv) and (wv+8)
#pragma unroll
  for (int t2 = 0; t2 < 2; ++t2) {
    const int dt = wv + t2 * 8;
    v8f acc = {};
    const __bf16* prow = &Pb[l16][0];
    const __bf16* vcol = V + ((size_t)(dt * 16 + l16) << 10) + hi * 16;
    for (int ks = 0; ks < 32; ++ks) {
      const int kb = ks * 32 + hi * 16;
      Frag pa;
      pa.h[0] = *(const bf16x8*)(prow + kb);
      pa.h[1] = *(const bf16x8*)(prow + kb + 8);
      v16bf vb = *(const v16bf*)(vcol + ks * 32);
      acc = __builtin_amdgcn_wmma_f32_16x16x32_bf16(
          false, pa.v, false, vb, (short)0, acc, false, false);
    }
#pragma unroll
    for (int r = 0; r < 8; ++r) {
      const int m    = r + 8 * hi;
      const int dcol = dt * 16 + l16;
      const unsigned int idx =
          ((unsigned)((h * 16 + b) * 1024 + q0 + m) << 8) + (unsigned)dcol;
      tempb[idx] = (__bf16)acc[r];
    }
  }
}

// ---------------------------------------------------------------------------
// Kernel 3: out = temp(bf16)[16384x1024] @ Wo_bf16^T + bo  -> f32 d_out
// grid (16, 64), block 256. Wave: 32 rows x 64 cols.
// ---------------------------------------------------------------------------
__global__ __launch_bounds__(256) void out_gemm_kernel(
    const __bf16* __restrict__ A, const __bf16* __restrict__ W,
    const float* __restrict__ bias, float* __restrict__ out) {
  const int tid  = threadIdx.x;
  const int wv   = tid >> 5;
  const int lane = tid & 31;
  const int l16  = lane & 15;
  const int hi   = lane >> 4;
  const int m0   = blockIdx.y * 256 + wv * 32;
  const int n0   = blockIdx.x * 64;

  const __bf16* arow0 = A + (size_t)(m0 + l16) * 1024;
  const __bf16* arow1 = arow0 + 16 * 1024;

  v8f acc0[4] = {};
  v8f acc1[4] = {};
  for (int kt = 0; kt < 32; ++kt) {
    const int kb  = kt * 32 + hi * 16;
    const int kbb = kt * 32 + hi * 16;
    __builtin_prefetch(arow0 + kb + 256, 0, 0);
    Frag a0, a1;
    a0.h[0] = *(const bf16x8*)(arow0 + kb);
    a0.h[1] = *(const bf16x8*)(arow0 + kb + 8);
    a1.h[0] = *(const bf16x8*)(arow1 + kb);
    a1.h[1] = *(const bf16x8*)(arow1 + kb + 8);
#pragma unroll
    for (int j = 0; j < 4; ++j) {
      v16bf bf = *(const v16bf*)(W + (size_t)(n0 + 16 * j + l16) * 1024 + kbb);
      acc0[j] = __builtin_amdgcn_wmma_f32_16x16x32_bf16(
          false, a0.v, false, bf, (short)0, acc0[j], false, false);
      acc1[j] = __builtin_amdgcn_wmma_f32_16x16x32_bf16(
          false, a1.v, false, bf, (short)0, acc1[j], false, false);
    }
  }
#pragma unroll
  for (int half = 0; half < 2; ++half) {
    const v8f* acc = half ? acc1 : acc0;
    const int mb = m0 + half * 16;
#pragma unroll
    for (int j = 0; j < 4; ++j) {
#pragma unroll
      for (int r = 0; r < 8; ++r) {
        const int mrow = mb + r + 8 * hi;
        const int ncol = n0 + 16 * j + l16;
        out[(size_t)mrow * 1024 + ncol] = acc[j][r] + bias[ncol];
      }
    }
  }
}

// ---------------------------------------------------------------------------
extern "C" void kernel_launch(void* const* d_in, const int* in_sizes, int n_in,
                              void* d_out, int out_size, void* d_ws, size_t ws_size,
                              hipStream_t stream) {
  const float* x   = (const float*)d_in[0];
  const float* y   = (const float*)d_in[1];
  const float* adj = (const float*)d_in[2];
  const float* Wq  = (const float*)d_in[3];
  const float* bq  = (const float*)d_in[4];
  const float* Wk  = (const float*)d_in[5];
  const float* bk  = (const float*)d_in[6];
  const float* Wv  = (const float*)d_in[7];
  const float* bv  = (const float*)d_in[8];
  const float* Wo  = (const float*)d_in[9];
  const float* bo  = (const float*)d_in[10];
  float* out = (float*)d_out;

  const size_t M16 = (size_t)16 * 1024 * 1024;  // 16M elements
  const size_t M1  = (size_t)1024 * 1024;       //  1M elements

  __bf16* xb    = (__bf16*)d_ws;
  __bf16* yb    = xb    + M16;
  __bf16* Wqb   = yb    + M16;
  __bf16* Wkb   = Wqb   + M1;
  __bf16* Wvb   = Wkb   + M1;
  __bf16* Wob   = Wvb   + M1;
  __bf16* Qb    = Wob   + M1;
  __bf16* Kb    = Qb    + M16;
  __bf16* Vtb   = Kb    + M16;
  __bf16* tempb = Vtb   + M16;

  dim3 blk(256);

  // 0) one-time bf16 conversion of activations + weights
  cvt_bf16_kernel<<<dim3((unsigned)(M16 / 8 / 256)), blk, 0, stream>>>(x,  xb,  (int)M16);
  cvt_bf16_kernel<<<dim3((unsigned)(M16 / 8 / 256)), blk, 0, stream>>>(y,  yb,  (int)M16);
  cvt_bf16_kernel<<<dim3((unsigned)(M1  / 8 / 256)), blk, 0, stream>>>(Wq, Wqb, (int)M1);
  cvt_bf16_kernel<<<dim3((unsigned)(M1  / 8 / 256)), blk, 0, stream>>>(Wk, Wkb, (int)M1);
  cvt_bf16_kernel<<<dim3((unsigned)(M1  / 8 / 256)), blk, 0, stream>>>(Wv, Wvb, (int)M1);
  cvt_bf16_kernel<<<dim3((unsigned)(M1  / 8 / 256)), blk, 0, stream>>>(Wo, Wob, (int)M1);

  // 1) QKV projections
  dim3 ggrid(16, 64);  // N/64, M/256
  qkv_gemm_kernel<<<ggrid, blk, 0, stream>>>(xb, Wqb, bq, Qb, 0);
  qkv_gemm_kernel<<<ggrid, blk, 0, stream>>>(yb, Wkb, bk, Kb, 0);
  qkv_gemm_kernel<<<ggrid, blk, 0, stream>>>(yb, Wvb, bv, Vtb, 1);

  // 2) fused attention (scores + softmax + PV), one block per (h,b,q-tile)
  attn_kernel<<<dim3(64, 16, 4), blk, 0, stream>>>(Qb, Kb, Vtb, adj, tempb);

  // 3) output projection (f32 result)
  out_gemm_kernel<<<ggrid, blk, 0, stream>>>(tempb, Wo != nullptr ? Wob : Wob, bo, out);
}
